// rnn_crf_86260123173477
// MI455X (gfx1250) — matrix operand order repaired and checked by reading, not compiled, not run on hardware
//
#include <hip/hip_runtime.h>
#include <hip/hip_bf16.h>
#include <math.h>

// ---------------------------------------------------------------------------
// CRF forward (logsumexp scan) for MI455X / gfx1250, wave32 + WMMA bf16
// + async tensor-path h staging (GLOBAL_LOAD_ASYNC_TO_LDS, ASYNCcnt).
//   state per batch b:  m[b] (log-offset), p[j,b] = exp(score[j,b] - m[b])
//   step:  V = E @ P  (E = exp(trans), constant);  ptmp_i = V_i * exp(h_i)
//          pmax = max_i ptmp;  p <- ptmp/pmax;  m <- m + log(pmax)
//   out_b = m + log( sum_j p_j * exp(trans[EOS, j]) )
// One wave owns 16 batch columns for the whole sequence (no barriers).
// h tiles are double-buffered in LDS via async loads issued one step ahead,
// with global_prefetch_b8 running 16 steps ahead to pre-stage L2/WGP$.
// ---------------------------------------------------------------------------

typedef __attribute__((ext_vector_type(16))) __bf16 v16bf;
typedef __attribute__((ext_vector_type(8)))  float  v8f;
typedef __attribute__((ext_vector_type(8)))  unsigned int v8u;

// int4 with gcc vector_size spelling to match the builtin's prototype exactly
typedef int vi4 __attribute__((vector_size(4 * sizeof(int))));
typedef vi4 __attribute__((address_space(1)))* gvi4p;   // global
typedef vi4 __attribute__((address_space(3)))* lvi4p;   // LDS

#define TAGS    64
#define SEQ     1024
#define BATCH   512
#define SOS_IDX 1
#define EOS_IDX 2
#define PDIST   16   // prefetch distance (steps)

// ---- CDNA5 async memory->LDS copy (ASYNCcnt) ------------------------------
#if defined(__has_builtin)
#if __has_builtin(__builtin_amdgcn_global_load_async_to_lds_b128)
#define ASYNC_B128(gp, lp)                                                    \
  __builtin_amdgcn_global_load_async_to_lds_b128(                             \
      (gvi4p)(unsigned long long)(size_t)(gp),                                \
      (lvi4p)(unsigned)(size_t)(lp), 0, 0)
#endif
#if __has_builtin(__builtin_amdgcn_s_wait_asynccnt)
#define WAIT_ASYNC(n) __builtin_amdgcn_s_wait_asynccnt(n)
#endif
#endif
#ifndef ASYNC_B128
#define ASYNC_B128(gp, lp)                                                    \
  asm volatile("global_load_async_to_lds_b128 %0, %1, off"                    \
               :: "v"((unsigned)(size_t)(lp)),                                \
                  "v"((unsigned long long)(size_t)(gp))                       \
               : "memory")
#endif
#ifndef WAIT_ASYNC
#define WAIT_ASYNC(n) asm volatile("s_wait_asynccnt " #n ::: "memory")
#endif

__global__ void crf_prep(const float* __restrict__ trans,
                         unsigned short* __restrict__ Ebf,  // [64*64] bf16 bits
                         float* __restrict__ Eend) {        // [64]
  int i = threadIdx.x;  // 0..63, row of trans
  for (int j = 0; j < TAGS; ++j) {
    float e = expf(trans[i * TAGS + j]);        // NEG rows/cols -> exactly 0
    __bf16 b = (__bf16)e;
    Ebf[i * TAGS + j] = __builtin_bit_cast(unsigned short, b);
  }
  Eend[i] = expf(trans[EOS_IDX * TAGS + i]);
}

__launch_bounds__(32)
__global__ void crf_fwd(const float* __restrict__ h,
                        const float* __restrict__ mask,
                        const unsigned short* __restrict__ Ebf,
                        const float* __restrict__ Eend,
                        float* __restrict__ out) {
  // Wave-private LDS tiles (one wave per block).
  __shared__ float        h_buf0[16 * 68];  // [b][i], stride 68 (bank padding)
  __shared__ float        h_buf1[16 * 68];  // double buffer
  __shared__ unsigned int p_lds[16 * 37];   // [b][pair j/2], stride 37 (odd)

  const int lane = threadIdx.x;      // 0..31
  const int b0   = blockIdx.x * 16;  // this wave's batch window
  const int bl   = lane & 15;        // batch column within window
  const int hi   = lane >> 4;        // 0: low half, 1: high half

  // ---- Build constant A-fragments of E (16-bit A layout, 16x32 per frag) ---
  // element e of v16bf: K_local = e + (hi ? 16 : 0), M = lane&15
  v16bf afrag[4][2];
  {
    const int dK = hi ? 16 : 0;
    #pragma unroll
    for (int mt = 0; mt < 4; ++mt)
      #pragma unroll
      for (int kc = 0; kc < 2; ++kc)
        #pragma unroll
        for (int e = 0; e < 16; ++e) {
          int K = kc * 32 + e + dK;
          unsigned short us = Ebf[(mt * 16 + bl) * TAGS + K];
          afrag[mt][kc][e] = __builtin_bit_cast(__bf16, us);
        }
  }

  // ---- Initial state: p = onehot(SOS), m = 0 ------------------------------
  // B-fragment element e of chunk kc holds row j = kc*32 + (hi?16:0) + e.
  v16bf pfrag[2];
  #pragma unroll
  for (int kc = 0; kc < 2; ++kc)
    #pragma unroll
    for (int e = 0; e < 16; ++e)
      pfrag[kc][e] = (__bf16)0.0f;
  if (hi == 0) pfrag[0][SOS_IDX] = (__bf16)1.0f;
  float moff = 0.0f;

  // ---- Async staging job for this lane: batch lb, 32-float half ------------
  const int lb = lane >> 1, half = lane & 1;
  const float* gsrc = h + ((size_t)(b0 + lb) * SEQ) * TAGS + half * 32;
  float* lbase[2] = {h_buf0 + lb * 68 + half * 32,
                     h_buf1 + lb * 68 + half * 32};

  // Prologue: async-copy tile 0 into buffer 0.
  {
    const float* g = gsrc;  // t = 0
    #pragma unroll
    for (int k = 0; k < 8; ++k) ASYNC_B128(g + 4 * k, lbase[0] + 4 * k);
  }

  for (int t = 0; t < SEQ; ++t) {
    // -- Issue async copy of tile t+1 (clamped) into the other buffer --------
    {
      const int tn = (t + 1 < SEQ) ? (t + 1) : (SEQ - 1);
      const float* g = gsrc + (size_t)tn * TAGS;
      float* l = lbase[(t + 1) & 1];
      #pragma unroll
      for (int k = 0; k < 8; ++k) ASYNC_B128(g + 4 * k, l + 4 * k);
    }
    // Deep prefetch into L2/WGP$ (one 128B line per lane per step).
    {
      const int tp = (t + PDIST < SEQ) ? (t + PDIST) : (SEQ - 1);
      __builtin_prefetch(gsrc + (size_t)tp * TAGS, 0, 1);
    }

    // -- V = E @ P : 4 M-tiles x 2 K-chunks of v_wmma_f32_16x16x32_bf16 -----
    // (independent of h; issues while async copy of tile t completes)
    v8f acc[4];
    #pragma unroll
    for (int mt = 0; mt < 4; ++mt) {
      v8f z = {0.f, 0.f, 0.f, 0.f, 0.f, 0.f, 0.f, 0.f};
      acc[mt] = __builtin_amdgcn_wmma_f32_16x16x32_bf16(
          false, afrag[mt][0], false, pfrag[0], (short)0, z, false, false);
      acc[mt] = __builtin_amdgcn_wmma_f32_16x16x32_bf16(
          false, afrag[mt][1], false, pfrag[1], (short)0, acc[mt], false, false);
    }

    // -- Wait for tile t (<=8 outstanding: only the 8 just-issued remain) ----
    WAIT_ASYNC(8);
    asm volatile("" ::: "memory");
    const float* hl = (t & 1) ? h_buf1 : h_buf0;

    // -- ptmp_i = V_i * exp(h_i); track per-batch max ------------------------
    // D layout: acc[mt][r] is (i = mt*16 + hi*8 + r, b = bl)
    float pmax = 0.0f;
    #pragma unroll
    for (int mt = 0; mt < 4; ++mt)
      #pragma unroll
      for (int r = 0; r < 8; ++r) {
        int i = mt * 16 + (hi ? 8 : 0) + r;
        float v = acc[mt][r] * __expf(hl[bl * 68 + i]);
        acc[mt][r] = v;
        pmax = fmaxf(pmax, v);
      }
    pmax = fmaxf(pmax, __shfl_xor(pmax, 16, 32));   // combine lane halves
    const float rs = __builtin_amdgcn_rcpf(pmax);

    // -- Normalize, pack bf16 pairs, write transpose tile to LDS -------------
    #pragma unroll
    for (int mt = 0; mt < 4; ++mt)
      #pragma unroll
      for (int s = 0; s < 4; ++s) {
        __bf16 blo = (__bf16)(acc[mt][2 * s]     * rs);
        __bf16 bhi = (__bf16)(acc[mt][2 * s + 1] * rs);
        unsigned int packed =
            ((unsigned int)__builtin_bit_cast(unsigned short, bhi) << 16) |
            (unsigned int)__builtin_bit_cast(unsigned short, blo);
        // pair index = i/2 = mt*8 + hi*4 + s
        p_lds[bl * 37 + mt * 8 + (hi ? 4 : 0) + s] = packed;
      }
    asm volatile("" ::: "memory");   // order writes before B-side reads

    // -- Mask blend + gather next B-fragments --------------------------------
    const float mk  = mask[(size_t)(b0 + bl) * SEQ + t];
    const bool  upd = (mk != 0.0f);
    const float mnew = moff + __logf(pmax);
    #pragma unroll
    for (int kc = 0; kc < 2; ++kc) {
      const unsigned int* pp = &p_lds[bl * 37 + kc * 16 + (hi ? 8 : 0)];
      v8u wv;
      #pragma unroll
      for (int q = 0; q < 8; ++q) wv[q] = pp[q];
      v16bf np = __builtin_bit_cast(v16bf, wv);
      if (upd) pfrag[kc] = np;
    }
    if (upd) moff = mnew;
  }

  // ---- Termination: out_b = m + log( sum_j p_j * Eend_j ) ------------------
  float s = 0.0f;
  #pragma unroll
  for (int kc = 0; kc < 2; ++kc)
    #pragma unroll
    for (int e = 0; e < 16; ++e) {
      int j = kc * 32 + (hi ? 16 : 0) + e;
      s += (float)pfrag[kc][e] * Eend[j];
    }
  s += __shfl_xor(s, 16, 32);
  if (hi == 0) out[b0 + bl] = moff + __logf(s);
}

extern "C" void kernel_launch(void* const* d_in, const int* in_sizes, int n_in,
                              void* d_out, int out_size, void* d_ws, size_t ws_size,
                              hipStream_t stream) {
  const float* h     = (const float*)d_in[0];   // (B,S,T) f32
  const float* mask  = (const float*)d_in[1];   // (B,S)   f32
  const float* trans = (const float*)d_in[2];   // (T,T)   f32
  float* out = (float*)d_out;                   // (B,)    f32

  unsigned short* Ebf = (unsigned short*)d_ws;                       // 8 KB
  float* Eend = (float*)((char*)d_ws + TAGS * TAGS * sizeof(unsigned short));

  crf_prep<<<1, TAGS, 0, stream>>>(trans, Ebf, Eend);
  crf_fwd<<<BATCH / 16, 32, 0, stream>>>(h, mask, Ebf, Eend, out);
}
